// XORContentAddressableMemory_60035052863706
// MI455X (gfx1250) — 8-bit matrix operand order repaired and checked by reading, compiled, not run on hardware
//
#include <hip/hip_runtime.h>
#include <hip/hip_bf16.h>
#include <stdint.h>

#define KEY_BITS   1024
#define VALUE_BITS 1024
#define CAPACITY   65536
#define BATCH      2048

#define KBYTES           KEY_BITS
#define KSTEPS           (KEY_BITS / 128)
#define MGROUPS          (BATCH / 128)
#define WAVES_PER_WG     8
#define KEY_CHUNKS       64
#define KEYS_PER_CHUNK   (CAPACITY / KEY_CHUNKS)
#define NTILES_PER_CHUNK (KEYS_PER_CHUNK / 16)
#define LDS_PITCH        1040
#define TILE_LDS_BYTES   (16 * LDS_PITCH)

typedef __attribute__((ext_vector_type(16))) int   v16i_t;
typedef __attribute__((ext_vector_type(8)))  float v8f_t;

__global__ __launch_bounds__(256)
void cam_pack_bits(const int* __restrict__ in,
                   unsigned int* __restrict__ out,
                   int* __restrict__ rowSums) {
    const int row  = blockIdx.x;
    const int t    = threadIdx.x;
    const int base = row * KEY_BITS + t * 4;

    const int b0 = in[base + 0] & 1;
    const int b1 = in[base + 1] & 1;
    const int b2 = in[base + 2] & 1;
    const int b3 = in[base + 3] & 1;

    unsigned int w = (b0 ? 0x38u : 0u)
                   | ((b1 ? 0x38u : 0u) << 8)
                   | ((b2 ? 0x38u : 0u) << 16)
                   | ((b3 ? 0x38u : 0u) << 24);
    out[row * (KEY_BITS / 4) + t] = w;

    if (rowSums != nullptr) {
        __shared__ int red[256];
        red[t] = b0 + b1 + b2 + b3;
        __syncthreads();
        for (int s = 128; s > 0; s >>= 1) {
            if (t < s) red[t] += red[t + s];
            __syncthreads();
        }
        if (t == 0) rowSums[row] = red[0];
    }
}

__device__ __forceinline__
void stage_tile_async(const unsigned char* gbase,
                      unsigned char* ldsbuf,
                      int tid) {
#pragma unroll
    for (int i = 0; i < 4; ++i) {
        const int e   = tid + i * 256;
        const int row = e >> 6;
        const int col = (e & 63) * 16;
        const unsigned int lds_addr =
            (unsigned int)(uintptr_t)(ldsbuf + row * LDS_PITCH + col);
        const unsigned int voff = (unsigned int)e * 16u;
        asm volatile("global_load_async_to_lds_b128 %0, %1, %2"
                     :
                     : "v"(lds_addr), "v"(voff),
                       "s"((unsigned long long)(uintptr_t)gbase)
                     : "memory");
    }
}

__device__ __forceinline__
void load_bfrag(v16i_t& bv, const unsigned char* p) {
#pragma unroll
    for (int c = 0; c < 8; ++c) {
        const int2 d = *(const int2*)(p + c * 16);
        bv[2 * c + 0] = d.x;
        bv[2 * c + 1] = d.y;
    }
}

__global__ __launch_bounds__(256)
void cam_wmma_score(const unsigned char* __restrict__ q8,
                    const unsigned char* __restrict__ k8,
                    const int* __restrict__ sumk,
                    unsigned long long* __restrict__ partials) {
    __shared__ unsigned char ldsB[2][TILE_LDS_BYTES];

    const int tid     = threadIdx.x;
    const int lane    = tid & 31;
    const int wave    = tid >> 5;
    const int chunk   = blockIdx.y;
    const int m0      = (blockIdx.x * WAVES_PER_WG + wave) * 16;
    const int key0    = chunk * KEYS_PER_CHUNK;
    const int halfSel = lane >> 4;
    const int lid     = lane & 15;

    v16i_t afrag[KSTEPS];
    {
        const unsigned char* qrow =
            q8 + (size_t)(m0 + lid) * KBYTES + halfSel * 8;
#pragma unroll
        for (int ts = 0; ts < KSTEPS; ++ts) {
            const unsigned char* p = qrow + ts * 128;
            v16i_t av;
#pragma unroll
            for (int c = 0; c < 8; ++c) {
                const int2 d = *(const int2*)(p + c * 16);
                av[2 * c + 0] = d.x;
                av[2 * c + 1] = d.y;
            }
            afrag[ts] = av;
        }
    }

    unsigned long long best[8];
#pragma unroll
    for (int j = 0; j < 8; ++j) best[j] = 0ull;

    stage_tile_async(k8 + (size_t)key0 * KBYTES, &ldsB[0][0], tid);

    for (int nt = 0; nt < NTILES_PER_CHUNK; ++nt) {
        const int cur = nt & 1;

        asm volatile("s_wait_asynccnt 0x0" ::: "memory");
        __syncthreads();

        if (nt + 1 < NTILES_PER_CHUNK) {
            stage_tile_async(
                k8 + (size_t)(key0 + (nt + 1) * 16) * KBYTES,
                &ldsB[cur ^ 1][0], tid);
        }

        const int keyBase = key0 + nt * 16;
        const int myKey   = keyBase + lid;
        const int sk      = sumk[myKey];
        const unsigned char* brow =
            &ldsB[cur][0] + lid * LDS_PITCH + halfSel * 8;

        v16i_t bv[2];
        v8f_t  acc = {};
        load_bfrag(bv[0], brow);
#pragma unroll
        for (int ts = 0; ts < KSTEPS; ++ts) {
            if (ts + 1 < KSTEPS)
                load_bfrag(bv[(ts + 1) & 1], brow + (ts + 1) * 128);
            acc = __builtin_amdgcn_wmma_f32_16x16x128_fp8_fp8(
                afrag[ts], bv[ts & 1], (short)0, acc, false, false);
        }

#pragma unroll
        for (int j = 0; j < 8; ++j) {
            const int score = 2 * (int)acc[j] - sk;
            const unsigned long long packed =
                ((unsigned long long)(unsigned)(score + 1024) << 32)
              | (unsigned)(CAPACITY - 1 - myKey);
            if (packed > best[j]) best[j] = packed;
        }
    }

#pragma unroll
    for (int j = 0; j < 8; ++j) {
        unsigned long long v = best[j];
        for (int m = 1; m <= 8; m <<= 1) {
            const unsigned long long o = __shfl_xor(v, m, 32);
            if (o > v) v = o;
        }
        best[j] = v;
    }

    if (lid == 0) {
#pragma unroll
        for (int j = 0; j < 8; ++j) {
            const int row = m0 + halfSel * 8 + j;
            partials[(size_t)row * KEY_CHUNKS + chunk] = best[j];
        }
    }
}

__global__ __launch_bounds__(256)
void cam_reduce_gather(const unsigned long long* __restrict__ partials,
                       const float* __restrict__ values,
                       float* __restrict__ out) {
    const int q = blockIdx.x;
    const int t = threadIdx.x;
    __shared__ int s_idx;

    if (t < 32) {
        unsigned long long v = partials[(size_t)q * KEY_CHUNKS + t];
        const unsigned long long v2 = partials[(size_t)q * KEY_CHUNKS + t + 32];
        if (v2 > v) v = v2;
        for (int m = 1; m < 32; m <<= 1) {
            const unsigned long long o = __shfl_xor(v, m, 32);
            if (o > v) v = o;
        }
        if (t == 0) s_idx = (CAPACITY - 1) - (int)(v & 0xffffffffu);
    }
    __syncthreads();

    const int idx = s_idx;
    const float4* src = (const float4*)values + (size_t)idx * (VALUE_BITS / 4);
    float4*       dst = (float4*)out          + (size_t)q   * (VALUE_BITS / 4);
    dst[t] = src[t];
}

extern "C" void kernel_launch(void* const* d_in, const int* in_sizes, int n_in,
                              void* d_out, int out_size, void* d_ws, size_t ws_size,
                              hipStream_t stream) {
    const int*   query  = (const int*)d_in[0];
    const int*   keys   = (const int*)d_in[1];
    const float* values = (const float*)d_in[2];
    float* out = (float*)d_out;

    char* ws = (char*)d_ws;
    size_t off = 0;
    unsigned char* k8 = (unsigned char*)(ws + off);
    off += (size_t)CAPACITY * KBYTES;
    unsigned char* q8 = (unsigned char*)(ws + off);
    off += (size_t)BATCH * KBYTES;
    int* sumk = (int*)(ws + off);
    off += (size_t)CAPACITY * sizeof(int);
    unsigned long long* partials = (unsigned long long*)(ws + off);

    cam_pack_bits<<<CAPACITY, 256, 0, stream>>>(keys,  (unsigned int*)k8, sumk);
    cam_pack_bits<<<BATCH,    256, 0, stream>>>(query, (unsigned int*)q8, nullptr);

    dim3 grid(MGROUPS, KEY_CHUNKS);
    cam_wmma_score<<<grid, 256, 0, stream>>>(q8, k8, sumk, partials);

    cam_reduce_gather<<<BATCH, 256, 0, stream>>>(partials, values, out);
}
